// GraphSAGE_11751030522721
// MI455X (gfx1250) — matrix operand order repaired and checked by reading, not compiled, not run on hardware
//
#include <hip/hip_runtime.h>
#include <hip/hip_bf16.h>
#include <math.h>

typedef _Float16 h16;
typedef __attribute__((ext_vector_type(16))) _Float16 v16h;
typedef __attribute__((ext_vector_type(8)))  float    v8f;
typedef __attribute__((ext_vector_type(2)))  _Float16 h16x2;

#define N0C 1000000
#define N1C 102400
#define N2C 10240
#define N3C 1024
#define FANOUT 10
#define INFEA 128
#define HID 256
#define NCLS 47
#define NCLS_PAD 48

// ---------------- zero fill ----------------
__global__ void zero_f32(float* __restrict__ p, int n) {
  int i = blockIdx.x * blockDim.x + threadIdx.x;
  if (i < n) p[i] = 0.0f;
}

// ---------------- edge scatter: agg[dst] += H[src], deg[dst] += 1 ----------------
template <typename T>
__global__ void scatter_accum(const T* __restrict__ H, const int* __restrict__ src,
                              const int* __restrict__ dst, float* __restrict__ agg,
                              float* __restrict__ deg, int nedges, int K) {
  int e    = (blockIdx.x * blockDim.x + threadIdx.x) >> 5;
  int lane = threadIdx.x & 31;
  if (e >= nedges) return;
  int s = src[e], d = dst[e];
  const T* hs = H + (size_t)s * K;
  float*   ad = agg + (size_t)d * K;
  for (int f = lane; f < K; f += 32)
    atomicAdd(&ad[f], (float)hs[f]);
  if (lane == 0) atomicAdd(&deg[d], 1.0f);
}

// ---------------- build concatenated f16 A = [self | mean_neigh] ----------------
template <typename TS>
__global__ void build_A(const TS* __restrict__ Hself, const float* __restrict__ agg,
                        const float* __restrict__ deg, h16* __restrict__ A,
                        int M, int Ks, int Kn) {
  int i = blockIdx.x;
  if (i >= M) return;
  int K = Ks + Kn;
  float dinv = 1.0f / fmaxf(deg[i], 1.0f);
  for (int k = threadIdx.x; k < Ks; k += blockDim.x)
    A[(size_t)i * K + k] = (h16)((float)Hself[(size_t)i * Ks + k]);
  for (int k = threadIdx.x; k < Kn; k += blockDim.x)
    A[(size_t)i * K + Ks + k] = (h16)(agg[(size_t)i * Kn + k] * dinv);
}

// ---------------- weight prep: Wt[n,k] = f16(concat(Ws,Wn)[k,n]), pad n>=N with 0 ----------------
__global__ void prep_W(const float* __restrict__ Ws, const float* __restrict__ Wn,
                       h16* __restrict__ Wt, int Khalf, int N, int Npad) {
  int K = 2 * Khalf;
  int idx = blockIdx.x * blockDim.x + threadIdx.x;
  if (idx >= Npad * K) return;
  int n = idx / K, k = idx - n * K;
  float v = 0.0f;
  if (n < N) v = (k < Khalf) ? Ws[(size_t)k * N + n] : Wn[(size_t)(k - Khalf) * N + n];
  Wt[idx] = (h16)v;
}

// ---------------- WMMA GEMM: C[M,NC] = A[M,KC] * Bt[NC,KC]^T, f16 in / f32 acc ----------------
// Compile-time KC/NC so all fragment loads and C stores use immediate offsets.
// One wave computes a 16 x (16*NW) tile via v_wmma_f32_16x16x32_f16.
// A frag (16-bit A 16x32, ISA 7.12.2): lane half hi: v<4 -> k = hi*8 + 2v ; v>=4 -> k = 16 + hi*8 + 2(v-4)
// B frag (16-bit B 32x16):             n = lane%16 ;  same k mapping as A frag
// C/D layout: VGPR j -> row = hi*8 + j, col = lane%16
template <int NW, int KC, int NC>
__global__ __launch_bounds__(128) void gemm_wmma(const h16* __restrict__ A,
                                                 const h16* __restrict__ Bt,
                                                 float* __restrict__ C, int M) {
  int lane  = threadIdx.x;
  int tileM = blockIdx.x * blockDim.y + threadIdx.y;
  if (tileM * 16 >= M) return;
  int colbase = blockIdx.y * (16 * NW);
  int r  = lane & 15;
  int hi = lane >> 4;
  const h16* arow  = A  + (size_t)(tileM * 16 + r) * KC;
  const h16* bbase = Bt + (size_t)(colbase + r) * KC;

  v8f acc[NW] = {};

#pragma unroll ((KC == 256) ? 8 : 4)
  for (int k0 = 0; k0 < KC; k0 += 32) {
    __builtin_prefetch(arow + k0 + 128, 0, 1);   // global_prefetch_b8, A stream
    v16h a;
#pragma unroll
    for (int v = 0; v < 8; ++v) {
      int kk = (v < 4) ? (k0 + hi * 8 + 2 * v)
                       : (k0 + 16 + hi * 8 + 2 * (v - 4));
      h16x2 p = *(const h16x2*)(arow + kk);
      a[2 * v] = p[0]; a[2 * v + 1] = p[1];
    }
#pragma unroll
    for (int w = 0; w < NW; ++w) {
      v16h b;
#pragma unroll
      for (int v = 0; v < 8; ++v) {
        int kk = (v < 4) ? (k0 + hi * 8 + 2 * v)
                         : (k0 + 16 + hi * 8 + 2 * (v - 4));
        h16x2 p = *(const h16x2*)(bbase + (size_t)w * 16 * KC + kk);
        b[2 * v] = p[0]; b[2 * v + 1] = p[1];
      }
      acc[w] = __builtin_amdgcn_wmma_f32_16x16x32_f16(
          /*neg_a=*/false, a, /*neg_b=*/false, b,
          /*c_mod=*/(short)0, acc[w], /*reuse_a=*/false, /*reuse_b=*/false);
    }
  }

  float* crow = C + (size_t)(tileM * 16 + hi * 8) * NC + colbase + r;
#pragma unroll
  for (int w = 0; w < NW; ++w)
#pragma unroll
    for (int j = 0; j < 8; ++j)
      crow[j * NC + w * 16] = acc[w][j];
}

// ---------------- epilogue: relu(G + b), row L2-normalize, write f16 (N == HID == 256) ----------------
__global__ __launch_bounds__(256) void act_norm(const float* __restrict__ G,
                                                const float* __restrict__ b,
                                                h16* __restrict__ H, int M) {
  __shared__ float red[8];
  int i = blockIdx.x;
  int t = threadIdx.x;
  float v  = fmaxf(G[(size_t)i * HID + t] + b[t], 0.0f);
  float ss = v * v;
#pragma unroll
  for (int off = 16; off > 0; off >>= 1) ss += __shfl_down(ss, off, 32);
  if ((t & 31) == 0) red[t >> 5] = ss;
  __syncthreads();
  if (t < 8) {
    float s = red[t];
#pragma unroll
    for (int off = 4; off > 0; off >>= 1) s += __shfl_down(s, off, 8);
    if (t == 0) red[0] = s;
  }
  __syncthreads();
  float inv = 1.0f / (sqrtf(red[0]) + 1e-12f);
  H[(size_t)i * HID + t] = (h16)(v * inv);
}

// ---------------- final softmax over 47 classes (row stride NCLS_PAD in G) ----------------
__global__ void softmax47(const float* __restrict__ G, const float* __restrict__ b,
                          float* __restrict__ out, int M) {
  int i = blockIdx.x * blockDim.x + threadIdx.x;
  if (i >= M) return;
  const float* g = G + (size_t)i * NCLS_PAD;
  float mx = -1e30f;
  for (int c = 0; c < NCLS; ++c) mx = fmaxf(mx, g[c] + b[c]);
  float s = 0.0f;
  for (int c = 0; c < NCLS; ++c) s += expf(g[c] + b[c] - mx);
  float inv = 1.0f / s;
  for (int c = 0; c < NCLS; ++c)
    out[(size_t)i * NCLS + c] = expf(g[c] + b[c] - mx) * inv;
}

// ---------------- host orchestration ----------------
extern "C" void kernel_launch(void* const* d_in, const int* in_sizes, int n_in,
                              void* d_out, int out_size, void* d_ws, size_t ws_size,
                              hipStream_t stream) {
  const float* x    = (const float*)d_in[0];
  const int*   src0 = (const int*)d_in[1];
  const int*   dst0 = (const int*)d_in[2];
  const int*   src1 = (const int*)d_in[3];
  const int*   dst1 = (const int*)d_in[4];
  const int*   src2 = (const int*)d_in[5];
  const int*   dst2 = (const int*)d_in[6];
  const float* Ws0  = (const float*)d_in[7];
  const float* Wn0  = (const float*)d_in[8];
  const float* b0   = (const float*)d_in[9];
  const float* Ws1  = (const float*)d_in[10];
  const float* Wn1  = (const float*)d_in[11];
  const float* b1   = (const float*)d_in[12];
  const float* Ws2  = (const float*)d_in[13];
  const float* Wn2  = (const float*)d_in[14];
  const float* b2   = (const float*)d_in[15];

  // workspace layout (256B aligned)
  size_t off = 0;
  auto take = [&](size_t bytes) -> char* {
    char* p = (char*)d_ws + off;
    off += (bytes + 255) & ~(size_t)255;
    return p;
  };
  float* deg = (float*)take((size_t)N1C * 4);
  float* agg = (float*)take((size_t)N1C * INFEA * 4);   // reused per layer
  h16*   A   = (h16*)  take((size_t)N1C * (2 * INFEA) * 2);
  float* G   = (float*)take((size_t)N1C * HID * 4);
  h16*   H   = (h16*)  take((size_t)N1C * HID * 2);
  h16*   W0t = (h16*)  take((size_t)HID * (2 * INFEA) * 2);      // [256, 256]
  h16*   W1t = (h16*)  take((size_t)HID * (2 * HID) * 2);        // [256, 512]
  h16*   W2t = (h16*)  take((size_t)NCLS_PAD * (2 * HID) * 2);   // [48, 512]

  // weight prep
  prep_W<<<(HID * 2 * INFEA + 255) / 256, 256, 0, stream>>>(Ws0, Wn0, W0t, INFEA, HID, HID);
  prep_W<<<(HID * 2 * HID + 255) / 256, 256, 0, stream>>>(Ws1, Wn1, W1t, HID, HID, HID);
  prep_W<<<(NCLS_PAD * 2 * HID + 255) / 256, 256, 0, stream>>>(Ws2, Wn2, W2t, HID, NCLS, NCLS_PAD);

  // ---- layer 0 ----
  {
    int nE = N1C * FANOUT;
    zero_f32<<<(N1C * INFEA + 255) / 256, 256, 0, stream>>>(agg, N1C * INFEA);
    zero_f32<<<(N1C + 255) / 256, 256, 0, stream>>>(deg, N1C);
    scatter_accum<float><<<(nE * 32 + 255) / 256, 256, 0, stream>>>(x, src0, dst0, agg, deg, nE, INFEA);
    build_A<float><<<N1C, 256, 0, stream>>>(x, agg, deg, A, N1C, INFEA, INFEA);
    gemm_wmma<4, 256, 256><<<dim3(N1C / 64, HID / 64), dim3(32, 4), 0, stream>>>(A, W0t, G, N1C);
    act_norm<<<N1C, 256, 0, stream>>>(G, b0, H, N1C);
  }
  // ---- layer 1 ----
  {
    int nE = N2C * FANOUT;
    zero_f32<<<(N2C * HID + 255) / 256, 256, 0, stream>>>(agg, N2C * HID);
    zero_f32<<<(N2C + 255) / 256, 256, 0, stream>>>(deg, N2C);
    scatter_accum<h16><<<(nE * 32 + 255) / 256, 256, 0, stream>>>(H, src1, dst1, agg, deg, nE, HID);
    build_A<h16><<<N2C, 256, 0, stream>>>(H, agg, deg, A, N2C, HID, HID);
    gemm_wmma<4, 512, 256><<<dim3(N2C / 64, HID / 64), dim3(32, 4), 0, stream>>>(A, W1t, G, N2C);
    act_norm<<<N2C, 256, 0, stream>>>(G, b1, H, N2C);  // H now holds layer-1 output
  }
  // ---- layer 2 ----
  {
    int nE = N3C * FANOUT;
    zero_f32<<<(N3C * HID + 255) / 256, 256, 0, stream>>>(agg, N3C * HID);
    zero_f32<<<(N3C + 255) / 256, 256, 0, stream>>>(deg, N3C);
    scatter_accum<h16><<<(nE * 32 + 255) / 256, 256, 0, stream>>>(H, src2, dst2, agg, deg, nE, HID);
    build_A<h16><<<N3C, 256, 0, stream>>>(H, agg, deg, A, N3C, HID, HID);
    gemm_wmma<3, 512, 48><<<dim3(N3C / 64, 1), dim3(32, 4), 0, stream>>>(A, W2t, G, N3C);
    softmax47<<<(N3C + 255) / 256, 256, 0, stream>>>(G, b2, (float*)d_out, N3C);
  }
  (void)in_sizes; (void)n_in; (void)out_size; (void)ws_size;
}
